// RWEncoder_12266426597455
// MI455X (gfx1250) — compile-verified, matrix operand order checked by reading, not run on hardware
//
#include <hip/hip_runtime.h>
#include <math.h>

// Problem constants (setup_inputs: N=4096 fixed; E derived from in_sizes).
#define NN 4096

typedef __attribute__((ext_vector_type(4)))  _Float16 v4h;
typedef __attribute__((ext_vector_type(8)))  _Float16 v8h;
typedef __attribute__((ext_vector_type(16))) _Float16 v16h;
typedef __attribute__((ext_vector_type(8)))  float    v8f;
typedef __attribute__((ext_vector_type(4)))  int      v4i;

union V16U { v16h v; v8h h[2]; };

// CDNA5 async global->LDS path (ASYNCcnt-tracked), guarded so compile never regresses.
#if defined(__HIP_DEVICE_COMPILE__) && __has_builtin(__builtin_amdgcn_global_load_async_to_lds_b128)
#define HAS_ASYNC_LDS 1
#else
#define HAS_ASYNC_LDS 0
#endif

static __device__ __forceinline__ void wait_asynccnt_zero() {
#if defined(__HIP_DEVICE_COMPILE__) && __has_builtin(__builtin_amdgcn_s_wait_asynccnt)
  __builtin_amdgcn_s_wait_asynccnt(0);
#elif defined(__HIP_DEVICE_COMPILE__) && HAS_ASYNC_LDS
  asm volatile("s_wait_asynccnt 0x0" ::: "memory");
#endif
}

// ---------------------------------------------------------------- utilities
__global__ void zero_f32_kernel(float* __restrict__ p, int n4) {
  int i = blockIdx.x * blockDim.x + threadIdx.x;
  int stride = gridDim.x * blockDim.x;
  float4* p4 = (float4*)p;
  float4 z = make_float4(0.f, 0.f, 0.f, 0.f);
  for (; i < n4; i += stride) p4[i] = z;
}

// deg[j] = sum of edge_weight over edges with col == j
__global__ void deg_kernel(const int* __restrict__ ei, const float* __restrict__ ew,
                           float* __restrict__ deg, int E) {
  int e = blockIdx.x * blockDim.x + threadIdx.x;
  if (e < E) atomicAdd(&deg[ei[E + e]], ew[e]);
}

// Mf32[row, col] += w / max(deg[col], eps)   (duplicates summed, as in reference)
__global__ void scatter_kernel(const int* __restrict__ ei, const float* __restrict__ ew,
                               const float* __restrict__ deg, float* __restrict__ Mf32, int E) {
  int e = blockIdx.x * blockDim.x + threadIdx.x;
  if (e < E) {
    int r = ei[e];
    int c = ei[E + e];
    float d = fmaxf(deg[c], 1e-12f);
    atomicAdd(&Mf32[(size_t)r * NN + c], ew[e] / d);
  }
}

__global__ void diag_from_f32(const float* __restrict__ Mf32, float* __restrict__ d, int N) {
  int i = blockIdx.x * blockDim.x + threadIdx.x;
  if (i < N) d[i] = Mf32[(size_t)i * N + i];
}

__global__ void cvt_kernel(const float* __restrict__ src, _Float16* __restrict__ dst, int n4) {
  int i = blockIdx.x * blockDim.x + threadIdx.x;
  int stride = gridDim.x * blockDim.x;
  const float4* s4 = (const float4*)src;
  for (; i < n4; i += stride) {
    float4 f = s4[i];
    v4h o = { (_Float16)f.x, (_Float16)f.y, (_Float16)f.z, (_Float16)f.w };
    *(v4h*)(dst + (size_t)i * 4) = o;
  }
}

// MT[j][i] = (f16)Mf32[i][j], LDS-tiled 64x64 (odd stride breaks bank conflicts)
__global__ void transpose_cvt_kernel(const float* __restrict__ src, _Float16* __restrict__ dst,
                                     int N) {
  __shared__ _Float16 tile[64][65];
  const int bx = blockIdx.x * 64;  // src col block
  const int by = blockIdx.y * 64;  // src row block
  const int tx = threadIdx.x & 63;
  const int ty = threadIdx.x >> 6;  // 0..3
#pragma unroll
  for (int r = 0; r < 64; r += 4)
    tile[r + ty][tx] = (_Float16)src[(size_t)(by + r + ty) * N + bx + tx];
  __syncthreads();
#pragma unroll
  for (int r = 0; r < 64; r += 4)
    dst[(size_t)(bx + r + ty) * N + by + tx] = tile[tx][r + ty];
}

// ---------------------------------------------------------------- WMMA GEMM
// Computes C = Aq x (M^T) with Aq, C row-major f16 and the B-operand streamed
// directly from row-major M (col n of M^T == row n of M -> K-contiguous loads,
// no transposes anywhere). f32 accumulate; diagonal written in f32.
// Block tile 128x128x32, 8 wave32 waves, each wave 32x64 (2x4 WMMA tiles).
// Double-buffered LDS; tiles filled with GLOBAL_LOAD_ASYNC_TO_LDS_B128 when available.
__global__ void __launch_bounds__(256)
gemm_kernel(const _Float16* __restrict__ Aq, const _Float16* __restrict__ Bm,
            _Float16* __restrict__ C, float* __restrict__ diagOut, int N) {
  __shared__ _Float16 sA[2][128 * 40];  // [row][k], stride 40 halves (80B, 16B-aligned)
  __shared__ _Float16 sB[2][128 * 40];  // [col][k] == rows of M

  const int tid  = threadIdx.x;
  const int lane = tid & 31;
  const int wave = tid >> 5;
  const int rowBase = blockIdx.y * 128;
  const int colBase = blockIdx.x * 128;
  const int waveRow = (wave & 3) * 32;   // 4 wave rows
  const int waveCol = (wave >> 2) * 64;  // 2 wave cols
  const int lm = lane & 15;
  const int kb = (lane < 16) ? 0 : 8;    // K base per ISA 16-bit A/B layout

  // Per-thread tile-fill coordinates: 128 rows x 32 halves as 16B chunks.
  const int lr  = tid >> 2;         // row 0..63 (+64 on second pass)
  const int lkc = (tid & 3) << 3;   // k chunk {0,8,16,24}

  auto issue_tile = [&](int buf, int k0) {
#if HAS_ASYNC_LDS
#pragma unroll
    for (int p = 0; p < 2; ++p) {
      int r = lr + p * 64;
      __builtin_amdgcn_global_load_async_to_lds_b128(
          (__attribute__((address_space(1))) v4i*)const_cast<_Float16*>(
              Aq + (size_t)(rowBase + r) * N + k0 + lkc),
          (__attribute__((address_space(3))) v4i*)(&sA[buf][r * 40 + lkc]), 0, 0);
      __builtin_amdgcn_global_load_async_to_lds_b128(
          (__attribute__((address_space(1))) v4i*)const_cast<_Float16*>(
              Bm + (size_t)(colBase + r) * N + k0 + lkc),
          (__attribute__((address_space(3))) v4i*)(&sB[buf][r * 40 + lkc]), 0, 0);
    }
#else
#pragma unroll
    for (int p = 0; p < 2; ++p) {
      int r = lr + p * 64;
      v8h va = *(const v8h*)(Aq + (size_t)(rowBase + r) * N + k0 + lkc);
      *(v8h*)(&sA[buf][r * 40 + lkc]) = va;
      v8h vb = *(const v8h*)(Bm + (size_t)(colBase + r) * N + k0 + lkc);
      *(v8h*)(&sB[buf][r * 40 + lkc]) = vb;
    }
#endif
  };

  v8f acc[2][4] = {};
  const int nIt = N / 32;

  issue_tile(0, 0);
  for (int it = 0; it < nIt; ++it) {
    const int buf = it & 1;
    wait_asynccnt_zero();   // this wave's tile chunks have landed in LDS
    __syncthreads();        // every wave's chunks have landed
    if (it + 1 < nIt) issue_tile(buf ^ 1, (it + 1) * 32);  // overlap DMA with WMMA

    // Fragment loads: per lane, K groups {kb..kb+7} and {kb+16..kb+23}
    V16U af[2], bf[4];
#pragma unroll
    for (int ti = 0; ti < 2; ++ti) {
      int r = waveRow + ti * 16 + lm;
      af[ti].h[0] = *(const v8h*)(&sA[buf][r * 40 + kb]);
      af[ti].h[1] = *(const v8h*)(&sA[buf][r * 40 + kb + 16]);
    }
#pragma unroll
    for (int tj = 0; tj < 4; ++tj) {
      int cc = waveCol + tj * 16 + lm;
      bf[tj].h[0] = *(const v8h*)(&sB[buf][cc * 40 + kb]);
      bf[tj].h[1] = *(const v8h*)(&sB[buf][cc * 40 + kb + 16]);
    }

#pragma unroll
    for (int ti = 0; ti < 2; ++ti)
#pragma unroll
      for (int tj = 0; tj < 4; ++tj)
        acc[ti][tj] = __builtin_amdgcn_wmma_f32_16x16x32_f16(
            false, af[ti].v, false, bf[tj].v, (short)0, acc[ti][tj], false, false);
    __syncthreads();  // all waves done reading buf before it is refilled
  }

  // Epilogue: C/D layout — lanes 0-15: N=lane, M=v; lanes 16-31: N=lane-16, M=8+v
#pragma unroll
  for (int ti = 0; ti < 2; ++ti) {
#pragma unroll
    for (int tj = 0; tj < 4; ++tj) {
      int gRow0 = rowBase + waveRow + ti * 16 + ((lane < 16) ? 0 : 8);
      int gCol  = colBase + waveCol + tj * 16 + lm;
#pragma unroll
      for (int v = 0; v < 8; ++v) {
        int gRow = gRow0 + v;
        float val = acc[ti][tj][v];
        C[(size_t)gRow * N + gCol] = (_Float16)val;
        if (gRow == gCol) diagOut[gRow] = val;  // f32 diagonal before f16 downcast
      }
    }
  }
}

// ---------------------------------------------------------------- standardize
// X = [ones, diag1..diag8]; z-score each column (ddof=0, std clamped to 1e-6).
__global__ void standardize_kernel(const float* __restrict__ diag, float* __restrict__ out, int N) {
  __shared__ float s1[256], s2[256];
  const int c = blockIdx.x;  // 0..8
  const int tid = threadIdx.x;
  float sum = 0.f, sq = 0.f;
  for (int i = tid; i < N; i += 256) {
    float x = (c == 0) ? 1.0f : diag[(size_t)(c - 1) * N + i];
    sum += x; sq += x * x;
  }
  s1[tid] = sum; s2[tid] = sq;
  __syncthreads();
  for (int s = 128; s > 0; s >>= 1) {
    if (tid < s) { s1[tid] += s1[tid + s]; s2[tid] += s2[tid + s]; }
    __syncthreads();
  }
  float mean = s1[0] / (float)N;
  float var  = s2[0] / (float)N - mean * mean;
  float sd   = fmaxf(sqrtf(fmaxf(var, 0.f)), 1e-6f);
  for (int i = tid; i < N; i += 256) {
    float x = (c == 0) ? 1.0f : diag[(size_t)(c - 1) * N + i];
    out[(size_t)i * 9 + c] = (x - mean) / sd;
  }
}

// ---------------------------------------------------------------- launcher
extern "C" void kernel_launch(void* const* d_in, const int* in_sizes, int n_in,
                              void* d_out, int out_size, void* d_ws, size_t ws_size,
                              hipStream_t stream) {
  (void)n_in; (void)out_size; (void)ws_size;
  const int N = NN;                       // fixed by setup_inputs (d_in[2] holds 4096)
  const int E = in_sizes[0] / 2;          // edge_index is (2, E) int32
  const int* ei = (const int*)d_in[0];
  const float* ew = (const float*)d_in[1];
  float* out = (float*)d_out;

  // Workspace layout (~128.2 MB):
  //   [0,64MB)    Mf32 (f32 build buffer) — later overlaid by Q0/Q1 f16 ping-pong
  //   [64,96MB)   Mf16  (row-major M; streamed as the M^T B-operand)
  //   [96,128MB)  MT    (f16 M^T, first A-operand)
  //   [128MB..)   deg (N f32), diag (8*N f32)
  char* ws = (char*)d_ws;
  float*    Mf32 = (float*)ws;
  _Float16* Q0   = (_Float16*)ws;
  _Float16* Q1   = (_Float16*)(ws + ((size_t)32 << 20));
  _Float16* Mf16 = (_Float16*)(ws + ((size_t)64 << 20));
  _Float16* MT   = (_Float16*)(ws + ((size_t)96 << 20));
  float*    deg  = (float*)(ws + ((size_t)128 << 20));
  float*    diag = (float*)(ws + ((size_t)128 << 20) + (size_t)N * sizeof(float));

  zero_f32_kernel<<<2048, 256, 0, stream>>>(Mf32, N * (N / 4));
  zero_f32_kernel<<<4, 256, 0, stream>>>(deg, N / 4);

  int eb = (E + 255) / 256;
  deg_kernel<<<eb, 256, 0, stream>>>(ei, ew, deg, E);
  scatter_kernel<<<eb, 256, 0, stream>>>(ei, ew, deg, Mf32, E);

  diag_from_f32<<<N / 256, 256, 0, stream>>>(Mf32, diag, N);            // t = 1
  cvt_kernel<<<2048, 256, 0, stream>>>(Mf32, Mf16, N * (N / 4));
  transpose_cvt_kernel<<<dim3(N / 64, N / 64), 256, 0, stream>>>(Mf32, MT, N);
  // Mf32 dead after this point; Q ping-pong overlays it.

  // Q_t = Q_{t-1} * M^T  (Q = P^T, so diag(Q_t) == diag(P_t)); Q_1 = M^T.
  dim3 g(N / 128, N / 128);
  const _Float16* cur = MT;
  _Float16* bufs[2] = {Q0, Q1};
  for (int t = 2; t <= 8; ++t) {
    _Float16* nxt = bufs[t & 1];
    gemm_kernel<<<g, 256, 0, stream>>>(cur, Mf16, nxt, diag + (size_t)(t - 1) * N, N);
    cur = nxt;
  }

  standardize_kernel<<<9, 256, 0, stream>>>(diag, out, N);
}
